// GMReader2ConvAverageReadout_86303072845938
// MI455X (gfx1250) — matrix operand order repaired and checked by reading, not compiled, not run on hardware
//
#include <hip/hip_runtime.h>
#include <hip/hip_bf16.h>

// ---------------------------------------------------------------------------
// GMReader2ConvAverageReadout for MI455X (gfx1250, wave32)
// Memory-bound on edge scatter (~3.5 GB traffic -> ~150us @ 23.3 TB/s).
// Dense N x K @ K x 128 GEMMs use v_wmma_f32_16x16x32_bf16 (f32 accumulate).
// ---------------------------------------------------------------------------

#define NNODES 131072
#define PPER   4096
#define BG     32
#define NEDGE  2097152
#define INDIM  64
#define HIDD   128
#define OUTC   16
#define EPSN   1e-5f
#define SLOPEK 0.01f

typedef __bf16 bf16_t;
typedef bf16_t v16bf __attribute__((ext_vector_type(16)));
typedef float  v8f   __attribute__((ext_vector_type(8)));

// ---------------- degree / norm prep ----------------------------------------

__global__ __launch_bounds__(256) void degree_kernel(
    const int* __restrict__ src, const int* __restrict__ dst,
    float* __restrict__ degS, float* __restrict__ degD)
{
    int e = blockIdx.x * 256 + threadIdx.x;
    atomicAdd(&degS[src[e]], 1.0f);
    atomicAdd(&degD[dst[e]], 1.0f);
}

__global__ __launch_bounds__(256) void rsqrt_kernel(float* __restrict__ d)
{
    int i = blockIdx.x * 256 + threadIdx.x;
    d[i] = rsqrtf(fmaxf(d[i], 1.0f));
}

// ---------------- edge message scatter --------------------------------------
// thread t handles 4 consecutive features of one edge. F = 4 << LOGTPE.
template<int LOGTPE>
__global__ __launch_bounds__(256) void edge_scatter(
    const float* __restrict__ X, const float* __restrict__ rsSrc,
    const float* __restrict__ ew, const int* __restrict__ srcI,
    const int* __restrict__ dstI, float* __restrict__ M)
{
    const int F = 4 << LOGTPE;
    unsigned t  = blockIdx.x * 256u + threadIdx.x;
    unsigned e  = t >> LOGTPE;
    unsigned f0 = (t & ((1u << LOGTPE) - 1u)) << 2;
    const int s = srcI[e], d = dstI[e];
    const float w = ew[e] * rsSrc[s];
    const float4 v = *(const float4*)(X + (size_t)s * F + f0);
    float* out = M + (size_t)d * F + f0;
    atomicAdd(out + 0, w * v.x);
    atomicAdd(out + 1, w * v.y);
    atomicAdd(out + 2, w * v.z);
    atomicAdd(out + 3, w * v.w);
}

// ---------------- WMMA GEMM: X[N,128] = (A[N,K] * rowScale) @ W[K,128] ------
// One 16x16 output tile per wave; 8 waves per 256-thread block.
__global__ __launch_bounds__(256) void gemm_rowscale_wmma(
    const float* __restrict__ A, const float* __restrict__ rowScale,
    const float* __restrict__ W, float* __restrict__ X, int K)
{
    const int lane    = threadIdx.x & 31;
    const int wave    = (blockIdx.x << 3) | (threadIdx.x >> 5);
    const int r0      = (wave >> 3) << 4;   // row tile (128 cols -> 8 col tiles)
    const int n0      = (wave & 7) << 4;    // col tile
    const int halfSel = lane >> 4;          // 0: lanes 0-15, 1: lanes 16-31
    const int sub     = lane & 15;
    const int kb      = halfSel << 3;       // A K-offset: 0 or 8

    const int aRow  = r0 + sub;
    const float s   = rowScale[aRow];
    const float* ap = A + (size_t)aRow * K + kb;
    const float* wp = W + n0 + sub;

    v8f acc = {0.f, 0.f, 0.f, 0.f, 0.f, 0.f, 0.f, 0.f};

    for (int k0 = 0; k0 < K; k0 += 32) {
        // A fragment (16-bit A layout: halves 0..7 -> K=kb+0..7, 8..15 -> K=kb+16..23)
        const float4 q0 = *(const float4*)(ap + k0 + 0);
        const float4 q1 = *(const float4*)(ap + k0 + 4);
        const float4 q2 = *(const float4*)(ap + k0 + 16);
        const float4 q3 = *(const float4*)(ap + k0 + 20);
        v16bf a;
        a[0]  = (bf16_t)(q0.x * s); a[1]  = (bf16_t)(q0.y * s);
        a[2]  = (bf16_t)(q0.z * s); a[3]  = (bf16_t)(q0.w * s);
        a[4]  = (bf16_t)(q1.x * s); a[5]  = (bf16_t)(q1.y * s);
        a[6]  = (bf16_t)(q1.z * s); a[7]  = (bf16_t)(q1.w * s);
        a[8]  = (bf16_t)(q2.x * s); a[9]  = (bf16_t)(q2.y * s);
        a[10] = (bf16_t)(q2.z * s); a[11] = (bf16_t)(q2.w * s);
        a[12] = (bf16_t)(q3.x * s); a[13] = (bf16_t)(q3.y * s);
        a[14] = (bf16_t)(q3.z * s); a[15] = (bf16_t)(q3.w * s);

        // B fragment (halves 0..7 -> K=8*halfSel+0..7, 8..15 -> K=8*halfSel+16..23; col = sub)
        const int kOffB = k0 + (halfSel << 3);
        v16bf b;
        #pragma unroll
        for (int h = 0; h < 16; ++h)
            b[h] = (bf16_t)wp[(size_t)(kOffB + (h & 7) + ((h >> 3) << 4)) * HIDD];

        acc = __builtin_amdgcn_wmma_f32_16x16x32_bf16(
                  false, a, false, b, (short)0, acc, false, false);
    }

    // D layout: VGPR v -> row r0 + 8*halfSel + v, col n0 + sub
    float* xp = X + (size_t)(r0 + (halfSel << 3)) * HIDD + n0 + sub;
    #pragma unroll
    for (int v = 0; v < 8; ++v)
        xp[(size_t)v * HIDD] = acc[v];
}

// ---------------- GraphNorm stats -------------------------------------------
// 512 blocks x 128 threads; each block covers 256 rows; thread = feature.
__global__ __launch_bounds__(128) void stats_kernel(
    const float* __restrict__ X, float* __restrict__ stats)
{
    const int f = threadIdx.x;
    const size_t base = (size_t)blockIdx.x * 256 * HIDD + f;
    float s = 0.f, s2 = 0.f;
    for (int i = 0; i < 256; ++i) {
        float v = X[base + (size_t)i * HIDD];
        s += v; s2 += v * v;
    }
    atomicAdd(&stats[f], s);
    atomicAdd(&stats[HIDD + f], s2);
}

__global__ __launch_bounds__(128) void finalize_stats(
    float* __restrict__ stats, const float* __restrict__ alpha)
{
    const int f = threadIdx.x;
    const float inv_n = 1.0f / (float)NNODES;
    float mu  = stats[f] * inv_n;
    float ex2 = stats[HIDD + f] * inv_n;
    float a   = alpha[f];
    float var = ex2 - 2.0f * a * mu * mu + a * a * mu * mu;
    stats[2 * HIDD + f] = a * mu;                 // alpha * mu
    stats[3 * HIDD + f] = rsqrtf(var + EPSN);     // inv std
}

// ---------------- normalize + LeakyReLU + per-graph mean readout ------------
// 512 blocks x 128 threads; 16 blocks per graph (256 rows each, graph-aligned).
__global__ __launch_bounds__(128) void norm_readout(
    float* __restrict__ X, const float* __restrict__ stats,
    const float* __restrict__ gamma, const float* __restrict__ beta,
    float* __restrict__ rOut)
{
    const int f   = threadIdx.x;
    const int g   = blockIdx.x >> 4;              // graph id
    const size_t row0 = (size_t)blockIdx.x * 256;
    const float amu = stats[2 * HIDD + f];
    const float inv = stats[3 * HIDD + f];
    const float gm  = gamma[f];
    const float bt  = beta[f];
    float accum = 0.f;
    for (int i = 0; i < 256; ++i) {
        size_t idx = (row0 + i) * HIDD + f;
        float v = (X[idx] - amu) * inv * gm + bt;
        v = (v > 0.f) ? v : SLOPEK * v;
        X[idx] = v;
        accum += v;
    }
    atomicAdd(&rOut[g * HIDD + f], accum * (1.0f / (float)PPER));
}

// ---------------- final classifier: [32,256] @ [256,16]^T -------------------
__global__ __launch_bounds__(512) void classifier(
    const float* __restrict__ r, const float* __restrict__ Wc,
    float* __restrict__ out)
{
    const int t = threadIdx.x;
    const int b = t >> 4;
    const int o = t & 15;
    float acc = 0.f;
    for (int k = 0; k < HIDD; ++k)
        acc += r[b * HIDD + k] * Wc[o * (2 * HIDD) + k];
    for (int k = 0; k < HIDD; ++k)
        acc += r[BG * HIDD + b * HIDD + k] * Wc[o * (2 * HIDD) + HIDD + k];
    out[b * OUTC + o] = acc;
}

// ---------------------------------------------------------------------------

extern "C" void kernel_launch(void* const* d_in, const int* in_sizes, int n_in,
                              void* d_out, int out_size, void* d_ws, size_t ws_size,
                              hipStream_t stream)
{
    const float* features = (const float*)d_in[0];
    const float* ew       = (const float*)d_in[1];
    const int*   src      = (const int*)  d_in[2];
    const int*   dst      = (const int*)  d_in[3];
    const float* W1       = (const float*)d_in[4];
    const float* W2       = (const float*)d_in[5];
    const float* Wc       = (const float*)d_in[6];
    const float* gamma1   = (const float*)d_in[7];
    const float* beta1    = (const float*)d_in[8];
    const float* alpha1   = (const float*)d_in[9];
    const float* gamma2   = (const float*)d_in[10];
    const float* beta2    = (const float*)d_in[11];
    const float* alpha2   = (const float*)d_in[12];
    float* out = (float*)d_out;

    // Workspace layout (floats):
    float* rs    = (float*)d_ws;                    // [2N]: rs_src | rs_dst
    float* rsS   = rs;
    float* rsD   = rs + NNODES;
    float* stats = rs + 2 * (size_t)NNODES;         // [512]
    float* r     = stats + 512;                     // [2*B*HID] r1 | r2
    float* m     = r + 2 * BG * HIDD;               // [N*HID] message buffer
    float* h     = m + (size_t)NNODES * HIDD;       // [N*HID] x1/h1/x2/h2

    const int gemmBlocks = (NNODES / 16) * (HIDD / 16) / 8;   // 8192

    // --- degrees -> rsqrt norms ---
    hipMemsetAsync(rs, 0, sizeof(float) * 2 * (size_t)NNODES, stream);
    degree_kernel<<<NEDGE / 256, 256, 0, stream>>>(src, dst, rsS, rsD);
    rsqrt_kernel<<<2 * NNODES / 256, 256, 0, stream>>>(rs);

    hipMemsetAsync(r, 0, sizeof(float) * 2 * BG * HIDD, stream);

    // --- layer 1: conv(64 -> 128) ---
    hipMemsetAsync(m, 0, sizeof(float) * (size_t)NNODES * INDIM, stream);
    edge_scatter<4><<<(NEDGE * 16) / 256, 256, 0, stream>>>(features, rsS, ew, src, dst, m);
    gemm_rowscale_wmma<<<gemmBlocks, 256, 0, stream>>>(m, rsD, W1, h, INDIM);
    hipMemsetAsync(stats, 0, sizeof(float) * 2 * HIDD, stream);
    stats_kernel<<<512, HIDD, 0, stream>>>(h, stats);
    finalize_stats<<<1, HIDD, 0, stream>>>(stats, alpha1);
    norm_readout<<<512, HIDD, 0, stream>>>(h, stats, gamma1, beta1, r);

    // --- layer 2: conv(128 -> 128) ---
    hipMemsetAsync(m, 0, sizeof(float) * (size_t)NNODES * HIDD, stream);
    edge_scatter<5><<<(NEDGE * 32) / 256, 256, 0, stream>>>(h, rsS, ew, src, dst, m);
    gemm_rowscale_wmma<<<gemmBlocks, 256, 0, stream>>>(m, rsD, W2, h, HIDD);
    hipMemsetAsync(stats, 0, sizeof(float) * 2 * HIDD, stream);
    stats_kernel<<<512, HIDD, 0, stream>>>(h, stats);
    finalize_stats<<<1, HIDD, 0, stream>>>(stats, alpha2);
    norm_readout<<<512, HIDD, 0, stream>>>(h, stats, gamma2, beta2, r + BG * HIDD);

    // --- classifier ---
    classifier<<<1, BG * OUTC, 0, stream>>>(r, Wc, out);
}
